// FRIENDATTN_67680094650650
// MI455X (gfx1250) — compile-verified, matrix operand order checked by reading, not run on hardware
//
#include <hip/hip_runtime.h>
#include <stdint.h>

typedef float      v2f  __attribute__((ext_vector_type(2)));
typedef float      v8f  __attribute__((ext_vector_type(8)));
typedef _Float16   v16h __attribute__((ext_vector_type(16)));
typedef int        v4i_g __attribute__((vector_size(16)));   // matches builtin param pointee

typedef __attribute__((address_space(1))) v4i_g* gp128_t;    // global int4*
typedef __attribute__((address_space(3))) v4i_g* lp128_t;    // LDS int4*

#define B_USERS 256
#define F_PER   64
#define L_HIST  50
#define D_DIM   128
#define N_ROWS  (B_USERS * F_PER)   // 16384
#define LROW    132                 // padded LDS row stride (floats): bank = (4l + d) % 64

__global__ __launch_bounds__(128)
void FRIENDATTN_kernel(const float* __restrict__ x,      // [N, L, D] row-major
                       const float* __restrict__ selfx,  // [B, D]
                       const unsigned char* __restrict__ mask, // [N, L] bool8
                       float* __restrict__ out)          // [N, D] (== [B,F,D] flat)
{
    __shared__ float sX[64 * LROW];  // 50 real rows + 14 zero rows (for 16-row WMMA tiles)
    __shared__ float sV[D_DIM];
    __shared__ float sS[64];         // raw scores
    __shared__ float sE[64];         // exp / final weights
    __shared__ float sRed[2];        // max, 1/sum

    const int n    = blockIdx.x;
    const int tid  = threadIdx.x;
    const int lane = tid & 31;
    const int wv   = tid >> 5;          // wave 0..3 -> M-tile rows 16*wv..16*wv+15
    const int hi   = (lane >> 4) & 1;   // half-wave selects K sub-pair
    const int mrow = lane & 15;         // M row within tile

    // ---- stage x[n] tile (50x128 f32) into padded LDS ----
#if __has_builtin(__builtin_amdgcn_global_load_async_to_lds_b128)
    {
        // gfx1250 async direct-to-LDS path (ASYNCcnt): no VGPR round-trip.
        const char* gbase = (const char*)(x + (size_t)n * (L_HIST * D_DIM));
        for (int q = tid; q < L_HIST * (D_DIM / 4); q += 128) {
            int l  = q >> 5;          // 32 x 16B per row
            int c4 = (q & 31) << 2;
            gp128_t gsrc = (gp128_t)(uintptr_t)(gbase + (size_t)q * 16);
            lp128_t ldst = (lp128_t)(unsigned int)(uintptr_t)&sX[l * LROW + c4];
            __builtin_amdgcn_global_load_async_to_lds_b128(gsrc, ldst, 0, 0);
        }
    }
#else
    {
        const float4* gx = (const float4*)(x + (size_t)n * (L_HIST * D_DIM));
        for (int q = tid; q < L_HIST * (D_DIM / 4); q += 128) {
            int l  = q >> 5;          // 32 float4 per row
            int c4 = (q & 31) << 2;
            float4 vv = gx[q];
            float* dst = &sX[l * LROW + c4];
            dst[0] = vv.x; dst[1] = vv.y; dst[2] = vv.z; dst[3] = vv.w;
        }
    }
#endif
    // zero pad rows 50..63 (only cols 0..127 are ever read) — overlaps with async fill
    for (int q = tid; q < (64 - L_HIST) * D_DIM; q += 128) {
        int l = L_HIST + (q >> 7);
        sX[l * LROW + (q & 127)] = 0.0f;
    }
    sV[tid] = selfx[(size_t)(n >> 6) * D_DIM + tid];

#if __has_builtin(__builtin_amdgcn_global_load_async_to_lds_b128)
  #if __has_builtin(__builtin_amdgcn_s_wait_asynccnt)
    __builtin_amdgcn_s_wait_asynccnt(0);
  #else
    asm volatile("s_wait_asynccnt 0x0" ::: "memory");
  #endif
#endif
    __syncthreads();

    // ---- scores via WMMA: S = X(16x128) . v, v broadcast into all 16 B columns ----
    const int r = (wv << 4) + mrow;
    v8f c = {};
#if __has_builtin(__builtin_amdgcn_wmma_f32_16x16x4_f32)
    // full-precision f32 path: 32 K-steps of 16x16x4
    #pragma unroll 8
    for (int k = 0; k < 32; ++k) {
        const int kb = (k << 2) + (hi << 1);
        v2f a; a.x = sX[r * LROW + kb]; a.y = sX[r * LROW + kb + 1];
        v2f b; b.x = sV[kb];            b.y = sV[kb + 1];
        c = __builtin_amdgcn_wmma_f32_16x16x4_f32(false, a, false, b,
                                                  (short)0, c, false, false);
    }
#else
    // fallback: codegen-confirmed f16 path, 4 K-chunks of 16x16x32
    #pragma unroll
    for (int kc = 0; kc < 4; ++kc) {
        const int kbase = kc << 5;
        v16h a, bf;
        #pragma unroll
        for (int p = 0; p < 8; ++p) {
            #pragma unroll
            for (int q = 0; q < 2; ++q) {
                const int kk = kbase + ((p >> 2) << 4) + ((p & 3) << 1) + (hi << 3) + q;
                a[2 * p + q]  = (_Float16)sX[r * LROW + kk];
                bf[2 * p + q] = (_Float16)sV[kk];
            }
        }
        c = __builtin_amdgcn_wmma_f32_16x16x32_f16(false, a, false, bf,
                                                   (short)0, c, false, false);
    }
#endif
    // C/D layout: VGPR j, lanes 0-15 -> M=j, lanes 16-31 -> M=j+8 (all N columns equal)
    if (mrow == 0) {
        union { v8f v; float f[8]; } cf; cf.v = c;
        const int base = (wv << 4) + (hi << 3);
        #pragma unroll
        for (int j = 0; j < 8; ++j) sS[base + j] = cf.f[j];
    }
    __syncthreads();

    // ---- softmax over L=50 (unmasked, matching reference), then mask weights ----
    if (tid == 0) {
        float m = sS[0];
        for (int l = 1; l < L_HIST; ++l) m = fmaxf(m, sS[l]);
        sRed[0] = m;
    }
    __syncthreads();
    if (tid < L_HIST) sE[tid] = __expf(sS[tid] - sRed[0]);
    __syncthreads();
    if (tid == 0) {
        float z = 0.0f;
        for (int l = 0; l < L_HIST; ++l) z += sE[l];
        sRed[1] = 1.0f / z;
    }
    __syncthreads();
    if (tid < L_HIST) {
        const float mk = mask[(size_t)n * L_HIST + tid] ? 1.0f : 0.0f;
        sE[tid] = sE[tid] * sRed[1] * mk;
    }
    __syncthreads();

    // ---- pooled[d] = sum_l w[l] * x[l][d], conflict-free LDS reads ----
    float acc = 0.0f;
    #pragma unroll 5
    for (int l = 0; l < L_HIST; ++l)
        acc += sE[l] * sX[l * LROW + tid];
    out[(size_t)n * D_DIM + tid] = acc;
}

extern "C" void kernel_launch(void* const* d_in, const int* in_sizes, int n_in,
                              void* d_out, int out_size, void* d_ws, size_t ws_size,
                              hipStream_t stream) {
    (void)in_sizes; (void)n_in; (void)out_size; (void)d_ws; (void)ws_size;
    const float*         x     = (const float*)d_in[0];
    const float*         selfx = (const float*)d_in[1];
    // d_in[2]=friend_num_src (scalar, uniform=64), d_in[3]=friend_num_src_tensor: unused
    const unsigned char* mask  = (const unsigned char*)d_in[4];
    float* out = (float*)d_out;

    FRIENDATTN_kernel<<<dim3(N_ROWS), dim3(128), 0, stream>>>(x, selfx, mask, out);
}
